// ProcessMappingGNN_77283641524344
// MI455X (gfx1250) — matrix operand order repaired and checked by reading, not compiled, not run on hardware
//
#include <hip/hip_runtime.h>

// MI455X / gfx1250 — wave32, WMMA + async global->LDS staging, pre-swizzled
// weights, CSR gather-aggregation (no feature atomics).
typedef __attribute__((ext_vector_type(16))) _Float16 v16h;
typedef __attribute__((ext_vector_type(8)))  _Float16 v8h;
typedef __attribute__((ext_vector_type(4)))  _Float16 v4h;
typedef __attribute__((ext_vector_type(2)))  _Float16 v2h;
typedef __attribute__((ext_vector_type(8)))  float    v8f;

union U16 { v16h w; v8h h[2]; };

#define NEG_SLOPE 0.2f

__device__ __forceinline__ float leaky(float x) { return x > 0.f ? x : NEG_SLOPE * x; }

// ---- CDNA5 async global->LDS copy (ASYNCcnt-tracked) ------------------------
__device__ __forceinline__ unsigned ldsOffset(const void* p) {
  return (unsigned)(unsigned long long)(__attribute__((address_space(3))) const void*)p;
}
__device__ __forceinline__ void async_copy_b128(unsigned ldsOff, const void* g) {
  asm volatile("global_load_async_to_lds_b128 %0, %1, off" :: "v"(ldsOff), "v"(g) : "memory");
}
__device__ __forceinline__ void async_wait0() {
  asm volatile("s_wait_asynccnt 0x0" ::: "memory");
}

// ---- One-shot weight pre-swizzle (f32 -> f16, WMMA-fragment order) ----------
__global__ void swizzle_B_kernel(const float* __restrict__ B, _Float16* __restrict__ out,
                                 int M, int mShift) {
  const int j = blockIdx.x * blockDim.x + threadIdx.x;   // (k2, m), k2 = k/2, K=128
  if (j >= 64 * M) return;
  const int m  = j & (M - 1);
  const int k2 = j >> mShift;
  const int nCT = M >> 4;
  const float b0 = B[(size_t)(2 * k2) * M + m];
  const float b1 = B[(size_t)(2 * k2 + 1) * M + m];
  const int chunk = k2 >> 4;
  const int kh    = (k2 >> 3) & 1;
  const int v     = k2 & 7;
  const int ct    = m >> 4;
  const int cc    = m & 15;
  const int lane  = kh * 16 + cc;
  const int off   = (((chunk * nCT + ct) * 32 + lane) << 4) + 2 * v;
  v2h pr = { (_Float16)b0, (_Float16)b1 };
  *(v2h*)&out[off] = pr;
}

// One wave: 16x16 tile, K=128 (4 WMMAs), 4 ds_load_b128 per WMMA.
__device__ __forceinline__ v8f wmma_row_k128(const _Float16* __restrict__ Arow,
                                             const _Float16* __restrict__ Bsw,
                                             int nCT, int ct, int lane, v8f c) {
  const int kh = lane >> 4;
  #pragma unroll
  for (int kb = 0; kb < 128; kb += 32) {
    const int chunk = kb >> 5;
    U16 a, b;
    a.h[0] = *(const v8h*)&Arow[kb + kh * 16];
    a.h[1] = *(const v8h*)&Arow[kb + kh * 16 + 8];
    const _Float16* Brun = &Bsw[((chunk * nCT + ct) * 32 + lane) << 4];
    b.h[0] = *(const v8h*)&Brun[0];
    b.h[1] = *(const v8h*)&Brun[8];
    c = __builtin_amdgcn_wmma_f32_16x16x32_f16(false, a.w, false, b.w, (short)0, c, false, false);
  }
  return c;
}

// ---- GEMM: C[nRows x M] (+opt f16 mirror) = act(A16[nRows x 128] @ B + bias)
template <bool HAS_C16, bool DO_RELU>
__global__ __launch_bounds__(256) void gemm_k128(
    const _Float16* __restrict__ A16, const _Float16* __restrict__ BswG,
    const float* __restrict__ bias, float* __restrict__ C, _Float16* __restrict__ C16,
    int nRows, int M)
{
  __shared__ __align__(16) _Float16 Bsw[128 * 128];  // 32 KB
  __shared__ __align__(16) _Float16 As[64 * 128];    // 16 KB
  const int tid  = threadIdx.x;
  const int nCT  = M >> 4;
  const int row0 = blockIdx.x * 64;

  for (int t = tid; t < 16 * M; t += 256)            // B: contiguous 128*M halfs
    async_copy_b128(ldsOffset(&Bsw[t * 8]), BswG + (size_t)t * 8);
  {
    const size_t lim = (size_t)nRows * 128 - 8;      // A: contiguous 16 KB (clamped tail)
    for (int t4 = tid; t4 < 64 * 16; t4 += 256) {
      size_t srcOff = (size_t)row0 * 128 + (size_t)t4 * 8;
      if (srcOff > lim) srcOff = lim;
      async_copy_b128(ldsOffset(&As[t4 * 8]), A16 + srcOff);
    }
  }
  async_wait0();
  __syncthreads();

  const int wave = tid >> 5, lane = tid & 31;
  if (wave < nCT) {
    const int r = lane & 15, kh = lane >> 4, cc = lane & 15;
    const int col = wave * 16 + cc;
    const float bval = bias ? bias[col] : 0.f;
    const bool fullTile = (row0 + 64 <= nRows);
    #pragma unroll
    for (int rt = 0; rt < 4; ++rt) {
      v8f c = {};
      c = wmma_row_k128(&As[(rt * 16 + r) * 128], Bsw, nCT, wave, lane, c);
      const int rbase = row0 + rt * 16 + 8 * kh;
      float*    Cp   = C + (size_t)rbase * M + col;
      _Float16* C16p = HAS_C16 ? C16 + (size_t)rbase * M + col : (_Float16*)0;
      if (fullTile) {
        #pragma unroll
        for (int v = 0; v < 8; ++v) {
          float val = c[v] + bval;
          if (DO_RELU) val = fmaxf(val, 0.f);
          *Cp = val; Cp += M;
          if (HAS_C16) { *C16p = (_Float16)val; C16p += M; }
        }
      } else {
        #pragma unroll
        for (int v = 0; v < 8; ++v) {
          if (rbase + v < nRows) {
            float val = c[v] + bval;
            if (DO_RELU) val = fmaxf(val, 0.f);
            Cp[(size_t)v * M] = val;
            if (HAS_C16) C16p[(size_t)v * M] = (_Float16)val;
          }
        }
      }
    }
  }
}

// ---- Fused edge head --------------------------------------------------------
__global__ __launch_bounds__(256) void edge_head_k256(
    const _Float16* __restrict__ h16, const int* __restrict__ src, const int* __restrict__ dst,
    const _Float16* __restrict__ We1swG, const float* __restrict__ be1,
    const float* __restrict__ We2, const float* __restrict__ be2,
    float* __restrict__ edge_pred, int E)
{
  __shared__ __align__(16) _Float16 Bsw[128 * 128];  // 32 KB
  __shared__ __align__(16) _Float16 As[64 * 128];    // 16 KB
  __shared__ float Hid[16 * 128];                    // 8 KB
  __shared__ float W2s[128 * 3];                     // 1.5 KB
  const int tid  = threadIdx.x;
  const int e0   = blockIdx.x * 64;
  const int wave = tid >> 5, lane = tid & 31;
  const int r = lane & 15, kh = lane >> 4, cc = lane & 15;

  if (tid < 128) {
    #pragma unroll
    for (int c2 = 0; c2 < 3; ++c2) W2s[c2 * 128 + tid] = We2[tid * 3 + c2];
  }

  v8f acc[4] = {};
  for (int chunk2 = 0; chunk2 < 2; ++chunk2) {
    __syncthreads();
    for (int t = tid; t < 16 * 128; t += 256)
      async_copy_b128(ldsOffset(&Bsw[t * 8]), We1swG + (size_t)chunk2 * 128 * 128 + (size_t)t * 8);
    for (int task = tid; task < 64 * 16; task += 256) {
      const int rr  = task >> 4;
      const int seg = task & 15;
      int e = e0 + rr; if (e >= E) e = E - 1;
      const int node = chunk2 ? dst[e] : src[e];
      async_copy_b128(ldsOffset(&As[rr * 128 + seg * 8]),
                      h16 + (size_t)node * 128 + seg * 8);
    }
    async_wait0();
    __syncthreads();
    #pragma unroll
    for (int t = 0; t < 4; ++t)
      acc[t] = wmma_row_k128(&As[(t * 16 + r) * 128], Bsw, 8, wave, lane, acc[t]);
  }

  const float b1v = be1[wave * 16 + cc];
  for (int t = 0; t < 4; ++t) {
    __syncthreads();
    #pragma unroll
    for (int v = 0; v < 8; ++v)
      Hid[(v + 8 * kh) * 128 + wave * 16 + cc] = fmaxf(acc[t][v] + b1v, 0.f);
    __syncthreads();
    if (tid < 48) {
      const int rr = tid / 3, col = tid % 3;
      const int e  = e0 + t * 16 + rr;
      if (e < E) {
        float s = be2[col];
        #pragma unroll 4
        for (int k = 0; k < 128; ++k) s += Hid[rr * 128 + k] * W2s[col * 128 + k];
        edge_pred[(size_t)e * 3 + col] = s;
      }
    }
  }
}

// ---- CSR construction (built once, reused by all layers) --------------------
__global__ void zero_int_kernel(int* __restrict__ p, int n) {
  const int i = blockIdx.x * blockDim.x + threadIdx.x;
  if (i < n) p[i] = 0;
}
__global__ void count_deg_kernel(const int* __restrict__ dst, int* __restrict__ deg, int E) {
  const int e = blockIdx.x * blockDim.x + threadIdx.x;
  if (e < E) atomicAdd(&deg[dst[e]], 1);
}
// per-block exclusive scan + block sums
__global__ void scan_block_kernel(const int* __restrict__ deg, int* __restrict__ exc,
                                  int* __restrict__ blockSums, int n) {
  __shared__ int tmp[256];
  const int i = blockIdx.x * 256 + threadIdx.x;
  const int v = (i < n) ? deg[i] : 0;
  tmp[threadIdx.x] = v;
  __syncthreads();
  for (int off = 1; off < 256; off <<= 1) {
    int t = 0;
    if (threadIdx.x >= off) t = tmp[threadIdx.x - off];
    __syncthreads();
    tmp[threadIdx.x] += t;
    __syncthreads();
  }
  if (i < n) exc[i] = tmp[threadIdx.x] - v;
  if (threadIdx.x == 255) blockSums[blockIdx.x] = tmp[255];
}
__global__ void scan_sums_serial_kernel(int* __restrict__ blockSums, int nb) {
  if (blockIdx.x == 0 && threadIdx.x == 0) {
    int run = 0;
    for (int i = 0; i < nb; ++i) { const int t = blockSums[i]; blockSums[i] = run; run += t; }
  }
}
__global__ void scan_add_kernel(int* __restrict__ rowptr, const int* __restrict__ blockSums,
                                int n, int Etot) {
  const int i = blockIdx.x * 256 + threadIdx.x;
  if (i < n) rowptr[i] += blockSums[i >> 8];
  if (i == 0) rowptr[n] = Etot;
}
__global__ void copy_int_kernel(const int* __restrict__ in, int* __restrict__ out, int n) {
  const int i = blockIdx.x * blockDim.x + threadIdx.x;
  if (i < n) out[i] = in[i];
}
__global__ void scatter_edges_kernel(const int* __restrict__ src, const int* __restrict__ dst,
                                     int* __restrict__ cursor, int* __restrict__ esrc, int E) {
  const int e = blockIdx.x * blockDim.x + threadIdx.x;
  if (e >= E) return;
  const int pos = atomicAdd(&cursor[dst[e]], 1);
  esrc[pos] = src[e];
}

// ---- elementwise f32 -> f16 (b128 in / b64 out) -----------------------------
__global__ void cvt16_kernel(const float* __restrict__ in, _Float16* __restrict__ out, size_t n4) {
  const size_t idx = (size_t)blockIdx.x * blockDim.x + threadIdx.x;
  if (idx >= n4) return;
  const float4 f = *(const float4*)&in[idx * 4];
  v4h hv = { (_Float16)f.x, (_Float16)f.y, (_Float16)f.z, (_Float16)f.w };
  *(v4h*)&out[idx * 4] = hv;
}

// ---- GAT per-layer kernels (gather-style, atomic-free) ----------------------
__global__ void node_alpha_kernel(const float* __restrict__ xw, const float* __restrict__ a_s,
                                  const float* __restrict__ a_d, float* __restrict__ os,
                                  float* __restrict__ od, int n) {
  const int gw   = (int)((blockIdx.x * blockDim.x + threadIdx.x) >> 5);
  const int lane = threadIdx.x & 31;
  if (gw >= n) return;
  float s = 0.f, d = 0.f;
  for (int k = lane; k < 128; k += 32) {
    const float v = xw[(size_t)gw * 128 + k];
    s += v * a_s[k];
    d += v * a_d[k];
  }
  #pragma unroll
  for (int off = 16; off > 0; off >>= 1) {
    s += __shfl_down(s, off, 32);
    d += __shfl_down(d, off, 32);
  }
  if (lane == 0) { os[gw] = s; od[gw] = d; }
}

// wave per node: softmax max & denom over incoming edges (+self), shuffle-reduced
__global__ void node_softmax_stats_kernel(const int* __restrict__ rowptr, const int* __restrict__ esrc,
                                          const float* __restrict__ as_, const float* __restrict__ ad_,
                                          float* __restrict__ mx, float* __restrict__ sm, int n) {
  const int node = (int)((blockIdx.x * blockDim.x + threadIdx.x) >> 5);
  const int lane = threadIdx.x & 31;
  if (node >= n) return;
  const int beg = rowptr[node], end = rowptr[node + 1];
  const float adv   = ad_[node];
  const float selfl = leaky(as_[node] + adv);
  float m = selfl;
  for (int j = beg + lane; j < end; j += 32) m = fmaxf(m, leaky(as_[esrc[j]] + adv));
  #pragma unroll
  for (int off = 16; off > 0; off >>= 1) m = fmaxf(m, __shfl_xor(m, off, 32));
  float s = (lane == 0) ? expf(selfl - m) : 0.f;
  for (int j = beg + lane; j < end; j += 32) s += expf(leaky(as_[esrc[j]] + adv) - m);
  #pragma unroll
  for (int off = 16; off > 0; off >>= 1) s += __shfl_xor(s, off, 32);
  if (lane == 0) { mx[node] = m; sm[node] = s; }
}

// wave per node: h = relu(h + sum_e alpha_e * xw[src_e] + alpha_self*xw[node] + b)
// each lane owns a contiguous float4 feature chunk; no atomics; fuses residual+relu+f16 mirror
__global__ void aggregate_fused_kernel(const float* __restrict__ xw,
                                       const int* __restrict__ rowptr, const int* __restrict__ esrc,
                                       const float* __restrict__ as_, const float* __restrict__ ad_,
                                       const float* __restrict__ mx, const float* __restrict__ sm,
                                       const float* __restrict__ b,
                                       float* __restrict__ h, _Float16* __restrict__ h16, int n) {
  const int node = (int)((blockIdx.x * blockDim.x + threadIdx.x) >> 5);
  const int lane = threadIdx.x & 31;
  if (node >= n) return;
  const int beg = rowptr[node], end = rowptr[node + 1];
  const float adv = ad_[node], m = mx[node], inv = 1.f / sm[node];

  const size_t base = (size_t)node * 128 + lane * 4;
  float4 acc;
  {
    const float wself = expf(leaky(as_[node] + adv) - m) * inv;
    const float4 xv = *(const float4*)&xw[base];
    acc = make_float4(wself * xv.x, wself * xv.y, wself * xv.z, wself * xv.w);
  }
  for (int j = beg; j < end; ++j) {
    const int s = esrc[j];
    const float w = expf(leaky(as_[s] + adv) - m) * inv;
    const float4 xv = *(const float4*)&xw[(size_t)s * 128 + lane * 4];
    acc.x += w * xv.x; acc.y += w * xv.y; acc.z += w * xv.z; acc.w += w * xv.w;
  }
  const float4 hv = *(const float4*)&h[base];
  const float4 bv = *(const float4*)&b[lane * 4];
  float4 r;
  r.x = fmaxf(hv.x + acc.x + bv.x, 0.f);
  r.y = fmaxf(hv.y + acc.y + bv.y, 0.f);
  r.z = fmaxf(hv.z + acc.z + bv.z, 0.f);
  r.w = fmaxf(hv.w + acc.w + bv.w, 0.f);
  *(float4*)&h[base] = r;
  v4h h4 = { (_Float16)r.x, (_Float16)r.y, (_Float16)r.z, (_Float16)r.w };
  *(v4h*)&h16[base] = h4;
}

// ---- graph embedding --------------------------------------------------------
__global__ void zero128_kernel(float* p) { if (threadIdx.x < 128) p[threadIdx.x] = 0.f; }

__global__ void colsum_kernel(const float* __restrict__ h, float* __restrict__ acc, int n) {
  const int col = threadIdx.x;  // blockDim = 128
  float s = 0.f;
  for (int r = blockIdx.x; r < n; r += gridDim.x) s += h[(size_t)r * 128 + col];
  atomicAdd(&acc[col], s);
}

__global__ void mean_kernel(const float* __restrict__ acc, float* __restrict__ out, int n) {
  if (threadIdx.x < 128) out[threadIdx.x] = acc[threadIdx.x] / (float)n;
}

extern "C" void kernel_launch(void* const* d_in, const int* in_sizes, int n_in,
                              void* d_out, int out_size, void* d_ws, size_t ws_size,
                              hipStream_t stream) {
  (void)n_in; (void)out_size; (void)ws_size;
  const float* x      = (const float*)d_in[0];
  const int*   ei     = (const int*)d_in[1];
  const float* W_emb  = (const float*)d_in[2];
  const float* b_emb  = (const float*)d_in[3];
  const float* gat_W  = (const float*)d_in[4];
  const float* gat_as = (const float*)d_in[5];
  const float* gat_ad = (const float*)d_in[6];
  const float* gat_b  = (const float*)d_in[7];
  const float* Wn1    = (const float*)d_in[8];
  const float* bn1    = (const float*)d_in[9];
  const float* Wn2    = (const float*)d_in[10];
  const float* bn2    = (const float*)d_in[11];
  const float* We1    = (const float*)d_in[12];
  const float* be1    = (const float*)d_in[13];
  const float* We2    = (const float*)d_in[14];
  const float* be2    = (const float*)d_in[15];

  const int N    = in_sizes[0] / 128;
  const int E    = in_sizes[1] / 2;
  const int L    = in_sizes[4] / (128 * 128);
  const int DOUT = in_sizes[10] / 128;
  const int* src = ei;
  const int* dst = ei + E;

  float*    h    = (float*)d_ws;
  float*    xw   = h    + (size_t)N * 128;
  _Float16* h16  = (_Float16*)(xw + (size_t)N * 128);         // N*128 halfs
  _Float16* t16  = h16 + (size_t)N * 128;                     // N*128 halfs (x16 / node-head)
  _Float16* wembS = t16 + (size_t)N * 128;                    // pre-swizzled weights (f16)
  _Float16* gatWS = wembS + 128 * 128;                        // L * 128*128
  _Float16* wn1S  = gatWS + (size_t)L * 128 * 128;
  _Float16* wn2S  = wn1S + 128 * 128;
  _Float16* we1S  = wn2S + 128 * 64;                          // 2 * 128*128
  float*    as_   = (float*)(we1S + 2 * 128 * 128);
  float*    ad_   = as_ + N;
  float*    mx    = ad_ + N;
  float*    sm    = mx  + N;
  float*    gacc  = sm  + N;
  int*      rowptr    = (int*)(gacc + 128);                   // N+1
  int*      cursor    = rowptr + (N + 1);                     // N (deg during count)
  int*      esrc      = cursor + N;                           // E
  int*      blockSums = esrc + E;                             // ceil(N/256)

  float* out       = (float*)d_out;
  float* node_pred = out;
  float* edge_pred = out + (size_t)N * DOUT;
  float* gemb      = edge_pred + (size_t)E * 3;

  const int rowBlocks = (N + 63) / 64;
  const int nodeWaveBlks = (int)(((size_t)N * 32 + 255) / 256);
  const int nScanBlks = (N + 255) / 256;
  const int eBlks = (E + 255) / 256;

  // ---- one-shot: pre-swizzle weights, convert x, build CSR ----
  swizzle_B_kernel<<<(64 * 128 + 255) / 256, 256, 0, stream>>>(W_emb, wembS, 128, 7);
  for (int l = 0; l < L; ++l)
    swizzle_B_kernel<<<(64 * 128 + 255) / 256, 256, 0, stream>>>(gat_W + (size_t)l * 128 * 128,
                                                                 gatWS + (size_t)l * 128 * 128, 128, 7);
  swizzle_B_kernel<<<(64 * 128 + 255) / 256, 256, 0, stream>>>(Wn1, wn1S, 128, 7);
  swizzle_B_kernel<<<(64 * 64 + 255) / 256, 256, 0, stream>>>(Wn2, wn2S, 64, 6);
  swizzle_B_kernel<<<(64 * 128 + 255) / 256, 256, 0, stream>>>(We1, we1S, 128, 7);
  swizzle_B_kernel<<<(64 * 128 + 255) / 256, 256, 0, stream>>>(We1 + 128 * 128, we1S + 128 * 128, 128, 7);
  cvt16_kernel<<<(int)(((size_t)N * 32 + 255) / 256), 256, 0, stream>>>(x, t16, (size_t)N * 32);

  zero_int_kernel<<<nScanBlks, 256, 0, stream>>>(cursor, N);
  count_deg_kernel<<<eBlks, 256, 0, stream>>>(dst, cursor, E);
  scan_block_kernel<<<nScanBlks, 256, 0, stream>>>(cursor, rowptr, blockSums, N);
  scan_sums_serial_kernel<<<1, 1, 0, stream>>>(blockSums, nScanBlks);
  scan_add_kernel<<<nScanBlks, 256, 0, stream>>>(rowptr, blockSums, N, E);
  copy_int_kernel<<<nScanBlks, 256, 0, stream>>>(rowptr, cursor, N);
  scatter_edges_kernel<<<eBlks, 256, 0, stream>>>(src, dst, cursor, esrc, E);

  // h = relu(x @ W_emb + b_emb) (f32 + f16 mirror)
  gemm_k128<true, true><<<rowBlocks, 256, 0, stream>>>(t16, wembS, b_emb, h, h16, N, 128);

  for (int l = 0; l < L; ++l) {
    gemm_k128<false, false><<<rowBlocks, 256, 0, stream>>>(h16, gatWS + (size_t)l * 128 * 128,
                                                           nullptr, xw, nullptr, N, 128);
    node_alpha_kernel<<<nodeWaveBlks, 256, 0, stream>>>(xw, gat_as + l * 128, gat_ad + l * 128, as_, ad_, N);
    node_softmax_stats_kernel<<<nodeWaveBlks, 256, 0, stream>>>(rowptr, esrc, as_, ad_, mx, sm, N);
    aggregate_fused_kernel<<<nodeWaveBlks, 256, 0, stream>>>(xw, rowptr, esrc, as_, ad_, mx, sm,
                                                             gat_b + l * 128, h, h16, N);
  }

  // node head
  gemm_k128<true, true><<<rowBlocks, 256, 0, stream>>>(h16, wn1S, bn1, xw, t16, N, 128);
  gemm_k128<false, false><<<rowBlocks, 256, 0, stream>>>(t16, wn2S, bn2, node_pred, nullptr, N, DOUT);

  // edge head (fused two-stage, async gathers from h16)
  edge_head_k256<<<(E + 63) / 64, 256, 0, stream>>>(h16, src, dst, we1S, be1, We2, be2, edge_pred, E);

  // graph embedding
  zero128_kernel<<<1, 128, 0, stream>>>(gacc);
  colsum_kernel<<<256, 128, 0, stream>>>(h, gacc, N);
  mean_kernel<<<1, 128, 0, stream>>>(gacc, gemb, N);
}
